// PointNetAE_455266533582
// MI455X (gfx1250) — compile-verified, matrix operand order checked by reading, not run on hardware
//
#include <hip/hip_runtime.h>
#include <hip/hip_bf16.h>
#include <stdint.h>

#define DEVFN __device__ __forceinline__

typedef __attribute__((ext_vector_type(16))) __bf16 bf16x16;
typedef __attribute__((ext_vector_type(8)))  float  f32x8;

#if defined(__has_builtin)
#if __has_builtin(__builtin_amdgcn_global_load_async_to_lds_b128) && \
    __has_builtin(__builtin_amdgcn_s_wait_asynccnt)
#define USE_ASYNC_LDS 1
#endif
#endif

#ifdef USE_ASYNC_LDS
typedef int v4i_t __attribute__((vector_size(16)));
typedef __attribute__((address_space(1))) v4i_t* gv4_p;   // global b128 ptr
typedef __attribute__((address_space(3))) v4i_t* lv4_p;   // LDS b128 ptr
#endif

union ABFrag { unsigned u[8]; bf16x16 v; };

// ---------------- scalar helpers ----------------
DEVFN unsigned short f2bf(float x) {                 // fp32 -> bf16 RNE
  unsigned u = __float_as_uint(x);
  u += 0x7FFFu + ((u >> 16) & 1u);
  return (unsigned short)(u >> 16);
}
DEVFN unsigned f2ord(float x) {                      // monotone float->uint map
  unsigned u = __float_as_uint(x);
  return (u & 0x80000000u) ? ~u : (u | 0x80000000u);
}
DEVFN float ord2f(unsigned u) {
  return (u & 0x80000000u) ? __uint_as_float(u & 0x7FFFFFFFu) : __uint_as_float(~u);
}

// ---------------- WMMA fragment load ----------------
// 16-bit A/B 16x32 layout (ISA 7.12.2): lane (16*half + r) holds row r,
// K = {half*8 + 0..7} in VGPR0-3 and {16 + half*8 + 0..7} in VGPR4-7.
DEVFN bf16x16 frag_load(const unsigned short* row, int kbase, int half) {
  const unsigned short* p = row + kbase + half * 8;
  ABFrag t;
  t.u[0] = *(const unsigned*)(p + 0);
  t.u[1] = *(const unsigned*)(p + 2);
  t.u[2] = *(const unsigned*)(p + 4);
  t.u[3] = *(const unsigned*)(p + 6);
  t.u[4] = *(const unsigned*)(p + 16);
  t.u[5] = *(const unsigned*)(p + 18);
  t.u[6] = *(const unsigned*)(p + 20);
  t.u[7] = *(const unsigned*)(p + 22);
  return t.v;
}

// One 16x16 D tile: A [16 x K] rows at A + r*lda, B = W^T with W rows [out][K].
DEVFN f32x8 gemm_tile(const unsigned short* A, int lda,
                      const unsigned short* Brow0, int ldb,
                      int K, float bias, int lane) {
  int half = lane >> 4, l15 = lane & 15;
  f32x8 acc = {bias, bias, bias, bias, bias, bias, bias, bias};
  const unsigned short* ar = A + (size_t)l15 * lda;
  const unsigned short* br = Brow0 + (size_t)l15 * ldb;
  for (int kb = 0; kb < K; kb += 32) {
    bf16x16 af = frag_load(ar, kb, half);
    bf16x16 bf = frag_load(br, kb, half);
    acc = __builtin_amdgcn_wmma_f32_16x16x32_bf16(false, af, false, bf,
                                                  (short)0, acc, false, false);
  }
  return acc;
}

// ---------------- sizes / workspace layout (bytes) ----------------
#define NB      8
#define NPTS    1024
#define NEXP    16
#define PADSLOT 1280           // 1024 + 16 experts * 15 pad
#define NTILE_E 80             // PADSLOT/16

static const size_t WS_W1T   = 0;                          // [16][256 out][256 in] bf16
static const size_t WS_W2T   = WS_W1T   + 16u*256*256*2;   // [16][128][256]
static const size_t WS_WC1   = WS_W2T   + 16u*128*256*2;   // [256][160] (136 padded)
static const size_t WS_WC2   = WS_WC1   + 256u*160*2;      // [512][256]
static const size_t WS_WC3   = WS_WC2   + 512u*256*2;      // [1024][512]
static const size_t WS_WD1T  = WS_WC3   + 1024u*512*2;     // [1024][1024]
static const size_t WS_WD2T  = WS_WD1T  + 1024u*1024*2;    // [2048][1024]
static const size_t WS_WD3T  = WS_WD2T  + 2048u*1024*2;    // [8192][2048]
static const size_t WS_SF    = WS_WD3T  + 8192u*2048*2;    // [8][1024][256] bf16
static const size_t WS_XI    = WS_SF    + 8u*1024*256*2;   // [8][1024][160] bf16
static const size_t WS_LATU  = WS_XI    + 8u*1024*160*2;   // [8][1024] u32
static const size_t WS_LAT16 = WS_LATU  + 8u*1024*4;       // [16][1024] bf16
static const size_t WS_D1    = WS_LAT16 + 16u*1024*2;      // [16][1024] bf16
static const size_t WS_D2    = WS_D1    + 16u*1024*2;      // [16][2048] bf16
static const size_t WS_CNT   = WS_D2    + 16u*2048*2;      // [8][16] i32
static const size_t WS_CUR   = WS_CNT   + 8u*16*4;
static const size_t WS_POFF  = WS_CUR   + 8u*16*4;
static const size_t WS_T2E   = WS_POFF  + 8u*16*4;         // [8][80] i32
static const size_t WS_PERM  = WS_T2E   + 8u*80*4;         // [8][1280] i32

// ---------------- utility kernels ----------------
__global__ void k_fill_u32(unsigned* p, unsigned v, int n) {
  int i = blockIdx.x * blockDim.x + threadIdx.x;
  if (i < n) p[i] = v;
}

__global__ void k_cvt(unsigned short* dst, const float* src, int n) {
  int i = blockIdx.x * blockDim.x + threadIdx.x;
  if (i < n) dst[i] = f2bf(src[i]);
}

// Tiled transpose-convert: src per-matrix [K][O] fp32 -> dst [O][K] bf16.
// K, O multiples of 32. Coalesced reads along o, coalesced writes along k.
__global__ void __launch_bounds__(256)
k_cvt_tr_tile(unsigned short* dst, const float* src, int K, int O) {
  __shared__ unsigned short tile[32][33];
  int mat = blockIdx.z;
  const float* s = src + (size_t)mat * K * O;
  unsigned short* d = dst + (size_t)mat * K * O;
  int k0 = blockIdx.y * 32, o0 = blockIdx.x * 32;
  for (int r = threadIdx.y; r < 32; r += 8)
    tile[r][threadIdx.x] = f2bf(s[(size_t)(k0 + r) * O + o0 + threadIdx.x]);
  __syncthreads();
  for (int r = threadIdx.y; r < 32; r += 8)
    d[(size_t)(o0 + r) * K + k0 + threadIdx.x] = tile[threadIdx.x][r];
}

// Wc1 [256 out][136 in] -> [256][160] bf16, zero-padded K
__global__ void k_cvt_pad_wc1(unsigned short* dst, const float* src) {
  int i = blockIdx.x * blockDim.x + threadIdx.x;
  if (i >= 256 * 160) return;
  int o = i / 160, k = i % 160;
  dst[i] = (k < 136) ? f2bf(src[o * 136 + k]) : (unsigned short)0;
}

// x[B][264][N] -> xi head channels (bf16, point-major) + sf (bf16, point-major)
// LDS-tiled transpose over (c, n); grid (N/32, ceil(264/32), B), block (32,8).
__global__ void __launch_bounds__(256)
k_prep_tile(const float* x, unsigned short* xi, unsigned short* sf) {
  __shared__ unsigned short tile[32][33];
  int b = blockIdx.z;
  int c0 = blockIdx.y * 32, n0 = blockIdx.x * 32;
  for (int r = threadIdx.y; r < 32; r += 8) {
    int c = c0 + r;
    unsigned short v = 0;
    if (c < 264) v = f2bf(x[((size_t)b * 264 + c) * NPTS + n0 + threadIdx.x]);
    tile[r][threadIdx.x] = v;
  }
  __syncthreads();
  for (int r = threadIdx.y; r < 32; r += 8) {
    int n = n0 + r;
    int c = c0 + threadIdx.x;
    if (c >= 264) continue;
    unsigned short v = tile[threadIdx.x][r];
    size_t pt = (size_t)b * NPTS + n;
    if (c < 8) xi[pt * 160 + c] = v;
    else       sf[pt * 256 + (c - 8)] = v;
  }
}

__global__ void k_xi_pad(unsigned short* xi) {  // zero channels 136..159
  int i = blockIdx.x * blockDim.x + threadIdx.x;
  if (i >= NB * NPTS * 24) return;
  int k = i % 24; int pt = i / 24;
  xi[(size_t)pt * 160 + 136 + k] = 0;
}

// ---------------- routing ----------------
__global__ void k_count(const int* cats, int* counts) {
  int i = blockIdx.x * blockDim.x + threadIdx.x;
  if (i >= NB * NPTS) return;
  atomicAdd(counts + (i >> 10) * NEXP + (cats[i] & 15), 1);
}

__global__ void k_route_plan(const int* counts, int* poff, int* t2e) {
  int b = threadIdx.x;
  if (b >= NB) return;
  int off = 0;
  for (int e = 0; e < NEXP; ++e) {
    poff[b * NEXP + e] = off;
    int pc = (counts[b * NEXP + e] + 15) >> 4;
    for (int t = 0; t < pc; ++t) t2e[b * NTILE_E + (off >> 4) + t] = e;
    off += pc << 4;
  }
  for (int t = off >> 4; t < NTILE_E; ++t) t2e[b * NTILE_E + t] = -1;
}

__global__ void k_scatter(const int* cats, const int* poff, int* curs, int* perm) {
  int i = blockIdx.x * blockDim.x + threadIdx.x;
  if (i >= NB * NPTS) return;
  int b = i >> 10, n = i & 1023, e = cats[i] & 15;
  int slot = poff[b * NEXP + e] + atomicAdd(curs + b * NEXP + e, 1);
  perm[b * PADSLOT + slot] = n;
}

// ---------------- expert MLP: 256 -> relu 256 -> 128, routed ----------------
__global__ void __launch_bounds__(128)
k_expert(const unsigned short* sf, unsigned short* xi,
         const unsigned short* W1t, const float* b1,
         const unsigned short* W2t, const float* b2,
         const int* perm, const int* t2e) {
  int b = blockIdx.x / NTILE_E, t = blockIdx.x % NTILE_E;
  int e = t2e[b * NTILE_E + t];
  if (e < 0) return;
  __shared__ __align__(16) unsigned short A0[16 * 256];
  __shared__ __align__(16) unsigned short H[16 * 256];
  __shared__ int pts[16];
  int tid = threadIdx.x, lane = tid & 31, wv = tid >> 5;
  int half = lane >> 4, l15 = lane & 15;
  if (tid < 16) pts[tid] = perm[b * PADSLOT + t * 16 + tid];
  __syncthreads();
  for (int i = tid; i < 16 * 128; i += 128) {       // 16 rows x 128 dwords
    int r = i >> 7, dw = i & 127;
    int p = pts[r];
    unsigned v = 0;
    if (p >= 0) v = ((const unsigned*)(sf + ((size_t)(b * NPTS + p)) * 256))[dw];
    ((unsigned*)A0)[r * 128 + dw] = v;
  }
  __syncthreads();
  const unsigned short* W1e = W1t + (size_t)e * 256 * 256;
  const float* b1e = b1 + e * 256;
  for (int nt = wv; nt < 16; nt += 4) {
    if (nt + 4 < 16) __builtin_prefetch(W1e + (size_t)(nt + 4) * 16 * 256, 0, 0);
    int n = nt * 16 + l15;
    f32x8 acc = gemm_tile(A0, 256, W1e + (size_t)nt * 16 * 256, 256, 256, b1e[n], lane);
#pragma unroll
    for (int j = 0; j < 8; ++j)
      H[(j + 8 * half) * 256 + n] = f2bf(fmaxf(acc[j], 0.f));
  }
  __syncthreads();
  const unsigned short* W2e = W2t + (size_t)e * 128 * 256;
  const float* b2e = b2 + e * 128;
  for (int nt = wv; nt < 8; nt += 4) {
    int n = nt * 16 + l15;
    f32x8 acc = gemm_tile(H, 256, W2e + (size_t)nt * 16 * 256, 256, 256, b2e[n], lane);
#pragma unroll
    for (int j = 0; j < 8; ++j) {
      int p = pts[j + 8 * half];
      if (p >= 0) xi[((size_t)(b * NPTS + p)) * 160 + 8 + n] = f2bf(acc[j]);
    }
  }
}

// ------- fused conv stack 160 -> 256 -> 512 -> 1024 + max-pool into latent -------
__global__ void __launch_bounds__(256)
k_conv(const unsigned short* xi,
       const unsigned short* Wc1, const float* bc1,
       const unsigned short* Wc2, const float* bc2,
       const unsigned short* Wc3, const float* bc3,
       unsigned* latu) {
  int b = blockIdx.x >> 6, t = blockIdx.x & 63;
  __shared__ __align__(16) unsigned short XI[16 * 160];
  __shared__ __align__(16) unsigned short H1[16 * 256];
  __shared__ __align__(16) unsigned short H2[16 * 512];
  int tid = threadIdx.x, lane = tid & 31, wv = tid >> 5;
  int half = lane >> 4, l15 = lane & 15;
  const unsigned short* srcp = xi + ((size_t)(b * NPTS + t * 16)) * 160;
#ifdef USE_ASYNC_LDS
  {
    // 16x160 bf16 tile = 5120 B = 320 x b128 async transfers, ASYNCcnt-tracked.
    unsigned lbase = (unsigned)(uintptr_t)(void*)XI;   // low 32 bits = LDS offset
    uintptr_t gbase = (uintptr_t)srcp;
    for (int i = tid; i < 320; i += 256)
      __builtin_amdgcn_global_load_async_to_lds_b128(
          (gv4_p)(gbase + (size_t)i * 16),
          (lv4_p)(uintptr_t)(lbase + i * 16), 0, 0);
    __builtin_amdgcn_s_wait_asynccnt(0);
  }
#else
  for (int i = tid; i < 16 * 80; i += 256)
    ((unsigned*)XI)[i] = ((const unsigned*)srcp)[i];
#endif
  __syncthreads();
  for (int nt = wv; nt < 16; nt += 8) {
    int n = nt * 16 + l15;
    f32x8 acc = gemm_tile(XI, 160, Wc1 + (size_t)nt * 16 * 160, 160, 160, bc1[n], lane);
#pragma unroll
    for (int j = 0; j < 8; ++j)
      H1[(j + 8 * half) * 256 + n] = f2bf(fmaxf(acc[j], 0.f));
  }
  __syncthreads();
  for (int nt = wv; nt < 32; nt += 8) {
    if (nt + 8 < 32) __builtin_prefetch(Wc2 + (size_t)(nt + 8) * 16 * 256, 0, 0);
    int n = nt * 16 + l15;
    f32x8 acc = gemm_tile(H1, 256, Wc2 + (size_t)nt * 16 * 256, 256, 256, bc2[n], lane);
#pragma unroll
    for (int j = 0; j < 8; ++j)
      H2[(j + 8 * half) * 512 + n] = f2bf(fmaxf(acc[j], 0.f));
  }
  __syncthreads();
  for (int nt = wv; nt < 64; nt += 8) {
    if (nt + 8 < 64) __builtin_prefetch(Wc3 + (size_t)(nt + 8) * 16 * 512, 0, 0);
    int n = nt * 16 + l15;
    f32x8 acc = gemm_tile(H2, 512, Wc3 + (size_t)nt * 16 * 512, 512, 512, bc3[n], lane);
    float mx = acc[0];
#pragma unroll
    for (int j = 1; j < 8; ++j) mx = fmaxf(mx, acc[j]);
    mx = fmaxf(mx, __shfl_xor(mx, 16, 32));          // combine the two lane-halves
    if (half == 0) atomicMax(latu + b * 1024 + n, f2ord(mx));
  }
}

// ---------------- latent decode + bf16 staging (rows 8..15 zero) ----------------
__global__ void k_latent_out(const unsigned* latu, float* lat_out, unsigned short* lat16) {
  int i = blockIdx.x * blockDim.x + threadIdx.x;
  if (i >= 16 * 1024) return;
  int r = i >> 10, c = i & 1023;
  if (r < 8) {
    float v = ord2f(latu[r * 1024 + c]);
    lat_out[r * 1024 + c] = v;
    lat16[i] = f2bf(v);
  } else lat16[i] = 0;
}

// ---------------- decoder GEMM (M=16), wave-per-N-tile ----------------
__global__ void __launch_bounds__(128)
k_dec_gemm(const unsigned short* A, int lda,
           const unsigned short* W, const float* bias, int K,
           unsigned short* out_bf, float* out_f32, int ldo) {
  int lane = threadIdx.x & 31, wv = threadIdx.x >> 5;
  int nt = blockIdx.x * 4 + wv;
  int half = lane >> 4, l15 = lane & 15;
  int n = nt * 16 + l15;
  __builtin_prefetch(W + (size_t)nt * 16 * K, 0, 0);
  f32x8 acc = gemm_tile(A, lda, W + (size_t)nt * 16 * K, K, K, bias[n], lane);
  if (out_bf) {
#pragma unroll
    for (int j = 0; j < 8; ++j)
      out_bf[(j + 8 * half) * ldo + n] = f2bf(fmaxf(acc[j], 0.f));
  } else {
#pragma unroll
    for (int j = 0; j < 8; ++j) { int m = j + 8 * half;
      if (m < 8) out_f32[(size_t)m * ldo + n] = acc[j]; }
  }
}

// ---------------- host launch ----------------
extern "C" void kernel_launch(void* const* d_in, const int* in_sizes, int n_in,
                              void* d_out, int out_size, void* d_ws, size_t ws_size,
                              hipStream_t stream) {
  (void)in_sizes; (void)n_in; (void)out_size; (void)ws_size;
  const float* x   = (const float*)d_in[0];
  const int*   cats= (const int*)d_in[1];
  const float* W1  = (const float*)d_in[2];  const float* b1  = (const float*)d_in[3];
  const float* W2  = (const float*)d_in[4];  const float* b2  = (const float*)d_in[5];
  const float* Wc1 = (const float*)d_in[6];  const float* bc1 = (const float*)d_in[7];
  const float* Wc2 = (const float*)d_in[8];  const float* bc2 = (const float*)d_in[9];
  const float* Wc3 = (const float*)d_in[10]; const float* bc3 = (const float*)d_in[11];
  const float* Wd1 = (const float*)d_in[12]; const float* bd1 = (const float*)d_in[13];
  const float* Wd2 = (const float*)d_in[14]; const float* bd2 = (const float*)d_in[15];
  const float* Wd3 = (const float*)d_in[16]; const float* bd3 = (const float*)d_in[17];

  char* ws = (char*)d_ws;
  unsigned short* w1t  = (unsigned short*)(ws + WS_W1T);
  unsigned short* w2t  = (unsigned short*)(ws + WS_W2T);
  unsigned short* wc1  = (unsigned short*)(ws + WS_WC1);
  unsigned short* wc2  = (unsigned short*)(ws + WS_WC2);
  unsigned short* wc3  = (unsigned short*)(ws + WS_WC3);
  unsigned short* wd1t = (unsigned short*)(ws + WS_WD1T);
  unsigned short* wd2t = (unsigned short*)(ws + WS_WD2T);
  unsigned short* wd3t = (unsigned short*)(ws + WS_WD3T);
  unsigned short* sf   = (unsigned short*)(ws + WS_SF);
  unsigned short* xi   = (unsigned short*)(ws + WS_XI);
  unsigned*       latu = (unsigned*)(ws + WS_LATU);
  unsigned short* lat16= (unsigned short*)(ws + WS_LAT16);
  unsigned short* d1   = (unsigned short*)(ws + WS_D1);
  unsigned short* d2   = (unsigned short*)(ws + WS_D2);
  int* cnt  = (int*)(ws + WS_CNT);
  int* cur  = (int*)(ws + WS_CUR);
  int* poff = (int*)(ws + WS_POFF);
  int* t2e  = (int*)(ws + WS_T2E);
  int* perm = (int*)(ws + WS_PERM);

  float* out = (float*)d_out;
  float* lat_out = out + NB * 8192;

  const int TB = 256;
  #define GRID(n) (((n) + TB - 1) / TB)

  // zero/init control state
  k_fill_u32<<<GRID(8 * 1024), TB, 0, stream>>>(latu, 0u, 8 * 1024);
  k_fill_u32<<<GRID(2 * 8 * 16), TB, 0, stream>>>((unsigned*)cnt, 0u, 2 * 8 * 16); // cnt+cur
  k_fill_u32<<<GRID(8 * PADSLOT), TB, 0, stream>>>((unsigned*)perm, 0xFFFFFFFFu, 8 * PADSLOT);

  // weight conversion (bf16, [out][in] layout) — LDS-tiled transposes
  dim3 tb32(32, 8);
  k_cvt_tr_tile<<<dim3(256/32, 256/32, 16), tb32, 0, stream>>>(w1t, W1, 256, 256);
  k_cvt_tr_tile<<<dim3(128/32, 256/32, 16), tb32, 0, stream>>>(w2t, W2, 256, 128);
  k_cvt_pad_wc1<<<GRID(256 * 160), TB, 0, stream>>>(wc1, Wc1);
  k_cvt<<<GRID(512 * 256), TB, 0, stream>>>(wc2, Wc2, 512 * 256);
  k_cvt<<<GRID(1024 * 512), TB, 0, stream>>>(wc3, Wc3, 1024 * 512);
  k_cvt_tr_tile<<<dim3(1024/32, 1024/32, 1), tb32, 0, stream>>>(wd1t, Wd1, 1024, 1024);
  k_cvt_tr_tile<<<dim3(2048/32, 1024/32, 1), tb32, 0, stream>>>(wd2t, Wd2, 1024, 2048);
  k_cvt_tr_tile<<<dim3(8192/32, 2048/32, 1), tb32, 0, stream>>>(wd3t, Wd3, 2048, 8192);

  // activation prep (tiled transpose) + xi channel padding
  k_prep_tile<<<dim3(NPTS/32, (264 + 31)/32, NB), tb32, 0, stream>>>(x, xi, sf);
  k_xi_pad<<<GRID(NB * NPTS * 24), TB, 0, stream>>>(xi);

  // routing
  k_count<<<GRID(NB * NPTS), TB, 0, stream>>>(cats, cnt);
  k_route_plan<<<1, 32, 0, stream>>>(cnt, poff, t2e);
  k_scatter<<<GRID(NB * NPTS), TB, 0, stream>>>(cats, poff, cur, perm);

  // expert MLP (WMMA)
  k_expert<<<NB * NTILE_E, 128, 0, stream>>>(sf, xi, w1t, b1, w2t, b2, perm, t2e);

  // fused conv stack + max-pool (WMMA, async-LDS staging)
  k_conv<<<NB * 64, 256, 0, stream>>>(xi, wc1, bc1, wc2, bc2, wc3, bc3, latu);

  // latent out + decoder (WMMA)
  k_latent_out<<<GRID(16 * 1024), TB, 0, stream>>>(latu, lat_out, lat16);
  k_dec_gemm<<<16, 128, 0, stream>>>(lat16, 1024, wd1t, bd1, 1024, d1, nullptr, 1024);
  k_dec_gemm<<<32, 128, 0, stream>>>(d1, 1024, wd2t, bd2, 1024, d2, nullptr, 2048);
  k_dec_gemm<<<128, 128, 0, stream>>>(d2, 2048, wd3t, bd3, 2048, nullptr, out, 8192);
  #undef GRID
}
